// VehicleActor_89060441850181
// MI455X (gfx1250) — matrix operand order repaired and checked by reading, not compiled, not run on hardware
//
#include <hip/hip_runtime.h>
#include <hip/hip_bf16.h>
#include <stdint.h>

// ---------------------------------------------------------------------------
// VehicleActor: LN -> Linear(64->512) -> 2-layer LSTM scanned over
// seq_len*B = 2048 sequential steps -> FC(512->8) + tanh.
//
// Phase 1 (parallel):   LN+pre linear; f32->f16 weight conversion;
//                       G0 = x @ W_ih0^T + b_ih0 + b_hh0 as a WMMA GEMM.
// Phase 2 (sequential): single 1024-thread workgroup runs the scan; the
//                       recurrent matvecs use WMMA with a broadcast-A trick
//                       (A rows all = h chunk from LDS, B = W^T tile from L2).
// ---------------------------------------------------------------------------

typedef _Float16 v8h  __attribute__((ext_vector_type(8)));
typedef _Float16 v16h __attribute__((ext_vector_type(16)));
typedef float    v8f  __attribute__((ext_vector_type(8)));

#define STATE_DIM 64
#define HID       512
#define FOURH     2048
#define ACT       8
#define BATCH     128

__device__ __forceinline__ float sigf(float x)  { return 1.0f / (1.0f + __expf(-x)); }

// ---- WMMA fragment loaders (CDNA5 16-bit A 16x32 / B 32x16 VGPR layouts) ----
// A (16x32, MxK): lane m=lane&15, hi=lane>>4.
//   halves 0..7  = A[m][kbase + 8*hi .. +7]        (VGPR0-3: K pairs)
//   halves 8..15 = A[m][kbase + 16 + 8*hi .. +7]   (VGPR4-7)
__device__ __forceinline__ v16h load_a_rows(const _Float16* A, int ld, int mbase,
                                            int kbase, int lane) {
  const int m  = mbase + (lane & 15);
  const int hi = lane >> 4;
  const _Float16* r = A + m * ld;
  v8h lo = *(const v8h*)(r + kbase + 8 * hi);
  v8h hh = *(const v8h*)(r + kbase + 16 + 8 * hi);
  v16h a;
#pragma unroll
  for (int i = 0; i < 8; ++i) { a[i] = lo[i]; a[i + 8] = hh[i]; }
  return a;
}

// A broadcast: every M row = the same vector h (so D rows are all identical,
// giving matvec结果 on M=0). Row index irrelevant; only hi selects K split.
__device__ __forceinline__ v16h load_a_bcast(const _Float16* h, int kbase, int lane) {
  const int hi = lane >> 4;
  v8h lo = *(const v8h*)(h + kbase + 8 * hi);
  v8h hh = *(const v8h*)(h + kbase + 16 + 8 * hi);
  v16h a;
#pragma unroll
  for (int i = 0; i < 8; ++i) { a[i] = lo[i]; a[i + 8] = hh[i]; }
  return a;
}

// B (32x16, KxN) with B[k][n] = W[nbase+n][k] (W row-major [rows][ld]):
//   lane n=lane&15, hi=lane>>4; halves 0..7 = W[nbase+n][kbase + 8*hi + e],
//   halves 8..15 = W[nbase+n][kbase + 16 + 8*hi + (e-8)]
__device__ __forceinline__ v16h load_b_wt(const _Float16* W, int ld, int nbase,
                                          int kbase, int lane) {
  const int n  = nbase + (lane & 15);
  const int hi = lane >> 4;
  const _Float16* r = W + n * ld + kbase + 8 * hi;
  v8h lo = *(const v8h*)(r);
  v8h hh = *(const v8h*)(r + 16);
  v16h b;
#pragma unroll
  for (int i = 0; i < 8; ++i) { b[i] = lo[i]; b[i + 8] = hh[i]; }
  return b;
}

// ---------------------------------------------------------------------------
// Kernel 1: f32 -> f16 conversion of LSTM weight matrices (layers 0 and 1).
// ---------------------------------------------------------------------------
__global__ void convert_w_kernel(const float* __restrict__ Wih,
                                 const float* __restrict__ Whh,
                                 _Float16* __restrict__ wih0,
                                 _Float16* __restrict__ whh0,
                                 _Float16* __restrict__ wih1,
                                 _Float16* __restrict__ whh1) {
  const int N = FOURH * HID;
  for (int i = blockIdx.x * blockDim.x + threadIdx.x; i < N;
       i += gridDim.x * blockDim.x) {
    wih0[i] = (_Float16)Wih[i];
    wih1[i] = (_Float16)Wih[N + i];
    whh0[i] = (_Float16)Whh[i];
    whh1[i] = (_Float16)Whh[N + i];
  }
}

// ---------------------------------------------------------------------------
// Kernel 2: LayerNorm(64) + x = xn @ W_pre^T + b_pre  -> X16 [128][512] f16
// One block per batch row, 64 threads.
// ---------------------------------------------------------------------------
__global__ void ln_pre_kernel(const float* __restrict__ state,
                              const float* __restrict__ ln_g,
                              const float* __restrict__ ln_b,
                              const float* __restrict__ W_pre,
                              const float* __restrict__ b_pre,
                              _Float16* __restrict__ X16) {
  __shared__ float sv[STATE_DIM];
  __shared__ float xn[STATE_DIM];
  const int b = blockIdx.x;
  const int t = threadIdx.x;
  sv[t] = state[b * STATE_DIM + t];
  __syncthreads();
  float mu = 0.f;
#pragma unroll
  for (int k = 0; k < STATE_DIM; ++k) mu += sv[k];
  mu *= (1.0f / STATE_DIM);
  float var = 0.f;
#pragma unroll
  for (int k = 0; k < STATE_DIM; ++k) { float d = sv[k] - mu; var += d * d; }
  var *= (1.0f / STATE_DIM);
  const float rs = __frsqrt_rn(var + 1e-5f);
  xn[t] = (sv[t] - mu) * rs * ln_g[t] + ln_b[t];
  __syncthreads();
  // 512 outputs / 64 threads = 8 each
#pragma unroll
  for (int j = 0; j < 8; ++j) {
    const int o = t * 8 + j;
    const float* w = W_pre + o * STATE_DIM;
    float acc = b_pre[o];
#pragma unroll
    for (int k = 0; k < STATE_DIM; ++k) acc += xn[k] * w[k];
    X16[b * HID + o] = (_Float16)acc;
  }
}

// ---------------------------------------------------------------------------
// Kernel 3: G0[128][2048] = X16 @ Wih0^T + b_ih0 + b_hh0   (WMMA GEMM)
// 1024 16x16 tiles; 8 waves/block, 128 blocks; K looped 512/32.
// ---------------------------------------------------------------------------
__global__ void gemm_g0_kernel(const _Float16* __restrict__ X16,
                               const _Float16* __restrict__ Wih0,
                               const float* __restrict__ b_ih,
                               const float* __restrict__ b_hh,
                               float* __restrict__ G0) {
  const int lane = threadIdx.x & 31;
  const int wid  = threadIdx.x >> 5;
  const int tile = blockIdx.x * 8 + wid;     // 0..1023
  const int mbase = (tile & 7) * 16;         // M = 128
  const int nbase = (tile >> 3) * 16;        // N = 2048
  v8f acc = {};
#pragma unroll
  for (int kc = 0; kc < HID / 32; ++kc) {
    v16h a = load_a_rows(X16, HID, mbase, kc * 32, lane);
    v16h bm = load_b_wt(Wih0, HID, nbase, kc * 32, lane);
    acc = __builtin_amdgcn_wmma_f32_16x16x32_f16(false, a, false, bm,
                                                 (short)0, acc, false, false);
  }
  const int hi  = lane >> 4;
  const int col = nbase + (lane & 15);
  const float bias = b_ih[col] + b_hh[col];
#pragma unroll
  for (int r = 0; r < 8; ++r) {
    const int m = mbase + r + 8 * hi;
    G0[m * FOURH + col] = acc[r] + bias;
  }
}

// ---------------------------------------------------------------------------
// Kernel 4: sequential 2-layer LSTM scan + FC head.
// Single 1024-thread workgroup (32 wave32 waves on one WGP).
// Recurrent matvec via WMMA broadcast-A: each wave owns 4 N-tiles (64 gates).
// ---------------------------------------------------------------------------
__global__ void
__launch_bounds__(1024, 1)
recurrent_kernel(const float* __restrict__ G0,
                 const _Float16* __restrict__ Whh0,
                 const _Float16* __restrict__ Wih1,
                 const _Float16* __restrict__ Whh1,
                 const float* __restrict__ b_ih,   // [2][2048]
                 const float* __restrict__ b_hh,   // [2][2048]
                 const float* __restrict__ W_fc,   // [8][512] f32
                 const float* __restrict__ b_fc,   // [8]
                 const int*   __restrict__ seq_len_p,
                 float* __restrict__ out) {        // [128][T*8]
  __shared__ float g[FOURH];
  __shared__ float hs0[HID], cs0[HID], hs1[HID], cs1[HID];
  __shared__ __align__(16) _Float16 h0_16[HID];
  __shared__ __align__(16) _Float16 h1_16[HID];

  const int tid  = threadIdx.x;
  const int lane = tid & 31;
  const int wid  = tid >> 5;                 // 0..31
  const int T    = seq_len_p[0];
  const int total = T * BATCH;

  if (tid < HID) {
    hs0[tid] = cs0[tid] = hs1[tid] = cs1[tid] = 0.f;
    h0_16[tid] = (_Float16)0.f;
    h1_16[tid] = (_Float16)0.f;
  }
  __syncthreads();

  const float* bih1 = b_ih + FOURH;
  const float* bhh1 = b_hh + FOURH;

  for (int s = 0; s < total; ++s) {
    const int brow = s & (BATCH - 1);        // s % 128
    const float* g0row = G0 + brow * FOURH;

    // ---- layer 0 recurrent matvec: g[n] = dot(Whh0[n], h0) ----
#pragma unroll
    for (int i = 0; i < 4; ++i) {
      const int ntile = wid * 4 + i;         // 0..127
      const int nbase = ntile * 16;
      __builtin_prefetch(Whh0 + (nbase + (lane & 15)) * HID, 0, 1);
      v8f acc = {};
#pragma unroll
      for (int kc = 0; kc < HID / 32; ++kc) {
        v16h a = load_a_bcast(h0_16, kc * 32, lane);
        v16h bm = load_b_wt(Whh0, HID, nbase, kc * 32, lane);
        acc = __builtin_amdgcn_wmma_f32_16x16x32_f16(false, a, false, bm,
                                                     (short)0, acc, false, false);
      }
      if (lane < 16) g[nbase + lane] = acc[0];   // M=0 row holds the matvec
    }
    __syncthreads();

    // ---- layer 0 gates (i,f,g,o) ----
    if (tid < HID) {
      const float gi = g0row[tid]            + g[tid];
      const float gf = g0row[HID + tid]      + g[HID + tid];
      const float gg = g0row[2 * HID + tid]  + g[2 * HID + tid];
      const float go = g0row[3 * HID + tid]  + g[3 * HID + tid];
      const float cn = sigf(gf) * cs0[tid] + sigf(gi) * tanhf(gg);
      const float hn = sigf(go) * tanhf(cn);
      cs0[tid] = cn; hs0[tid] = hn; h0_16[tid] = (_Float16)hn;
    }
    __syncthreads();

    // ---- layer 1: g[n] = dot(Wih1[n], h0) + dot(Whh1[n], h1) ----
#pragma unroll
    for (int i = 0; i < 4; ++i) {
      const int ntile = wid * 4 + i;
      const int nbase = ntile * 16;
      __builtin_prefetch(Wih1 + (nbase + (lane & 15)) * HID, 0, 1);
      v8f acc = {};
#pragma unroll
      for (int kc = 0; kc < HID / 32; ++kc) {
        v16h a = load_a_bcast(h0_16, kc * 32, lane);
        v16h bm = load_b_wt(Wih1, HID, nbase, kc * 32, lane);
        acc = __builtin_amdgcn_wmma_f32_16x16x32_f16(false, a, false, bm,
                                                     (short)0, acc, false, false);
      }
#pragma unroll
      for (int kc = 0; kc < HID / 32; ++kc) {
        v16h a = load_a_bcast(h1_16, kc * 32, lane);
        v16h bm = load_b_wt(Whh1, HID, nbase, kc * 32, lane);
        acc = __builtin_amdgcn_wmma_f32_16x16x32_f16(false, a, false, bm,
                                                     (short)0, acc, false, false);
      }
      if (lane < 16) g[nbase + lane] = acc[0];
    }
    __syncthreads();

    // ---- layer 1 gates ----
    if (tid < HID) {
      const float gi = g[tid]           + bih1[tid]           + bhh1[tid];
      const float gf = g[HID + tid]     + bih1[HID + tid]     + bhh1[HID + tid];
      const float gg = g[2 * HID + tid] + bih1[2 * HID + tid] + bhh1[2 * HID + tid];
      const float go = g[3 * HID + tid] + bih1[3 * HID + tid] + bhh1[3 * HID + tid];
      const float cn = sigf(gf) * cs1[tid] + sigf(gi) * tanhf(gg);
      const float hn = sigf(go) * tanhf(cn);
      cs1[tid] = cn; hs1[tid] = hn; h1_16[tid] = (_Float16)hn;
    }
    __syncthreads();

    // ---- FC head: 8 outputs, one wave each, lane-parallel reduction ----
    if (wid < ACT) {
      const float* w = W_fc + wid * HID;
      float p = 0.f;
#pragma unroll
      for (int k = lane; k < HID; k += 32) p += w[k] * hs1[k];
#pragma unroll
      for (int off = 16; off > 0; off >>= 1) p += __shfl_xor(p, off, 32);
      if (lane == 0) {
        const int tseq = s >> 7;             // s / 128
        out[brow * (T * ACT) + tseq * ACT + wid] = tanhf(p + b_fc[wid]);
      }
    }
    __syncthreads();
  }
}

// ---------------------------------------------------------------------------
extern "C" void kernel_launch(void* const* d_in, const int* in_sizes, int n_in,
                              void* d_out, int out_size, void* d_ws, size_t ws_size,
                              hipStream_t stream) {
  const float* state = (const float*)d_in[0];
  const float* ln_g  = (const float*)d_in[1];
  const float* ln_b  = (const float*)d_in[2];
  const float* W_pre = (const float*)d_in[3];
  const float* b_pre = (const float*)d_in[4];
  const float* W_ih  = (const float*)d_in[5];
  const float* W_hh  = (const float*)d_in[6];
  const float* b_ih  = (const float*)d_in[7];
  const float* b_hh  = (const float*)d_in[8];
  const float* W_fc  = (const float*)d_in[9];
  const float* b_fc  = (const float*)d_in[10];
  const int*   seqp  = (const int*)d_in[11];
  float* out = (float*)d_out;

  uint8_t* ws = (uint8_t*)d_ws;
  _Float16* X16    = (_Float16*)(ws);                       // 128*512*2   = 128 KB
  _Float16* wih0   = (_Float16*)(ws + (128u << 10));        // 2048*512*2  = 2 MB
  _Float16* whh0   = (_Float16*)(ws + (128u << 10) + 1u * (2048u * 512u * 2u));
  _Float16* wih1   = (_Float16*)(ws + (128u << 10) + 2u * (2048u * 512u * 2u));
  _Float16* whh1   = (_Float16*)(ws + (128u << 10) + 3u * (2048u * 512u * 2u));
  float*    G0     = (float*)   (ws + (128u << 10) + 4u * (2048u * 512u * 2u)); // 1 MB

  convert_w_kernel<<<1024, 256, 0, stream>>>(W_ih, W_hh, wih0, whh0, wih1, whh1);
  ln_pre_kernel<<<BATCH, STATE_DIM, 0, stream>>>(state, ln_g, ln_b, W_pre, b_pre, X16);
  gemm_g0_kernel<<<128, 256, 0, stream>>>(X16, wih0, b_ih, b_hh, G0);
  recurrent_kernel<<<1, 1024, 0, stream>>>(G0, whh0, wih1, whh1, b_ih, b_hh,
                                           W_fc, b_fc, seqp, out);
}
